// GraphGPSEncoder_77352361001431
// MI455X (gfx1250) — compile-verified, matrix operand order checked by reading, not run on hardware
//
#include <hip/hip_runtime.h>
#include <hip/hip_bf16.h>
#include <math.h>
#include <string.h>

#define NN   16384
#define EE   262144
#define HH   128
#define EINW 16
#define TT   10
#define DSZ  16
#define DCV  4
#define DTRK 8
#define NTT  (NN * TT)

typedef __attribute__((ext_vector_type(16))) __bf16 v16bf;
typedef __attribute__((ext_vector_type(8)))  float  v8f;

// ---------------------------------------------------------------- activations
__device__ __forceinline__ float actf(float x, int act) {
  if (act == 1) return fmaxf(x, 0.f);                                     // relu
  if (act == 2) return 0.5f * x * (1.f + erff(x * 0.70710678118654752f)); // gelu exact
  if (act == 3) return x / (1.f + __expf(-x));                            // silu
  if (act == 4) return (x > 20.f) ? x : log1pf(__expf(x));                // softplus
  return x;
}

// ------------------------------------- fp32 -> packed bf16 with zero padding
// dst is [Mp, Kp] (Kp multiple of 32, Mp multiple of 16); src is [rows, >=K]
// with row stride ld. Out-of-range (r >= rows || c >= K) -> 0.
__global__ void k_f32_to_bf16_pad(const float* __restrict__ src, int ld,
                                  unsigned short* __restrict__ dst,
                                  long Mp, int rows, int Kp, int K) {
  long i = (long)blockIdx.x * blockDim.x + threadIdx.x;
  if (i >= Mp * (long)Kp) return;
  long r = i / Kp; int c = (int)(i - r * (long)Kp);
  float f = (r < rows && c < K) ? src[r * (long)ld + c] : 0.f;
  unsigned int u = __float_as_uint(f);
  u += 0x7FFFu + ((u >> 16) & 1u);
  dst[i] = (unsigned short)(u >> 16);
}

// ----------------------------------------------------------- WMMA bf16 GEMM
// C[M,Nc] = act(A[M,Kp] @ W[Ncp,Kp]^T + bias). Operands are pre-padded packed
// bf16, so fragment loads are unconditional: per lane, a 16-bit fragment is
// two contiguous 8-element K-runs -> two 16-byte global_load_b128 each.
__global__ void k_gemm_wmma(const unsigned short* __restrict__ A,
                            const unsigned short* __restrict__ W,
                            const float* __restrict__ bias,
                            float* __restrict__ C,
                            int M, int Nc, int Kp, int act) {
  int gw   = (int)(((long)blockIdx.x * blockDim.x + threadIdx.x) >> 5);
  int lane = threadIdx.x & 31;
  int tn = (Nc + 15) >> 4;
  int tiles = (M >> 4) * tn;
  if (gw >= tiles) return;
  int tile_m = gw / tn, tile_n = gw % tn;
  int m0 = tile_m << 4, n0 = tile_n << 4;
  int half = lane >> 4, lidx = lane & 15;
  const unsigned short* ap = A + (long)(m0 + lidx) * Kp + (half << 3);
  const unsigned short* wp = W + (long)(n0 + lidx) * Kp + (half << 3);

  v8f acc;
#pragma unroll
  for (int r = 0; r < 8; ++r) acc[r] = 0.f;

  for (int k0 = 0; k0 < Kp; k0 += 32) {
    union { v16bf v; uint4 q[2]; } ua, uw;
    ua.q[0] = *(const uint4*)(ap + k0);        // K = k0 + half*8 + [0,8)
    ua.q[1] = *(const uint4*)(ap + k0 + 16);   // K = k0 + 16 + half*8 + [0,8)
    uw.q[0] = *(const uint4*)(wp + k0);
    uw.q[1] = *(const uint4*)(wp + k0 + 16);
    acc = __builtin_amdgcn_wmma_f32_16x16x32_bf16(false, ua.v, false, uw.v,
                                                  (short)0, acc, false, false);
  }
  int col = n0 + lidx;
  if (col < Nc) {
    float bv = bias ? bias[col] : 0.f;
#pragma unroll
    for (int r = 0; r < 8; ++r) {
      int row = m0 + r + (half << 3);          // C/D layout: vgpr r -> M = r + half*8
      C[(long)row * Nc + col] = actf(acc[r] + bv, act);
    }
  }
}

// ----------------------------------------------------------------- norms
__global__ void k_layernorm(const float* __restrict__ x, const float* __restrict__ g,
                            const float* __restrict__ b, float* __restrict__ y,
                            long M, int D, int act) {
  long row = (long)blockIdx.x * (blockDim.x >> 5) + (threadIdx.x >> 5);
  int lane = threadIdx.x & 31;
  if (row >= M) return;
  const float* xr = x + row * D;
  float s = 0.f;
  for (int c = lane; c < D; c += 32) s += xr[c];
  for (int o = 16; o > 0; o >>= 1) s += __shfl_xor(s, o, 32);
  float mean = s / D;
  float v = 0.f;
  for (int c = lane; c < D; c += 32) { float d = xr[c] - mean; v += d * d; }
  for (int o = 16; o > 0; o >>= 1) v += __shfl_xor(v, o, 32);
  float inv = rsqrtf(v / D + 1e-5f);
  float* yr = y + row * D;
  for (int c = lane; c < D; c += 32)
    yr[c] = actf((xr[c] - mean) * inv * g[c] + b[c], act);
}

__global__ void k_rmsnorm(const float* __restrict__ x, const float* __restrict__ w,
                          float* __restrict__ y, long M, int D, int act) {
  long row = (long)blockIdx.x * (blockDim.x >> 5) + (threadIdx.x >> 5);
  int lane = threadIdx.x & 31;
  if (row >= M) return;
  const float* xr = x + row * D;
  float s = 0.f;
  for (int c = lane; c < D; c += 32) { float v = xr[c]; s += v * v; }
  for (int o = 16; o > 0; o >>= 1) s += __shfl_xor(s, o, 32);
  float inv = rsqrtf(s / D + 1e-5f);
  float* yr = y + row * D;
  for (int c = lane; c < D; c += 32)
    yr[c] = actf(xr[c] * inv * w[c], act);
}

// ----------------------------------------------------------- simple fills/ew
__global__ void k_fillf(float* p, float v, long n) {
  long i = (long)blockIdx.x * blockDim.x + threadIdx.x;
  if (i < n) p[i] = v;
}
__global__ void k_fillu(unsigned int* p, unsigned int v, long n) {
  long i = (long)blockIdx.x * blockDim.x + threadIdx.x;
  if (i < n) p[i] = v;
}
__global__ void k_add(float* y, const float* x, long n) {
  long i = (long)blockIdx.x * blockDim.x + threadIdx.x;
  if (i < n) y[i] += x[i];
}
__global__ void k_mul_silu(float* y, const float* x, long n) {
  long i = (long)blockIdx.x * blockDim.x + threadIdx.x;
  if (i < n) { float v = x[i]; y[i] *= v / (1.f + __expf(-v)); }
}

// ---------------------------------------------------- GENConv segment softmax
__device__ __forceinline__ unsigned int fenc(float f) {
  unsigned int u = __float_as_uint(f);
  return (u & 0x80000000u) ? ~u : (u | 0x80000000u);
}
__device__ __forceinline__ float fdec(unsigned int u) {
  return (u & 0x80000000u) ? __uint_as_float(u & 0x7FFFFFFFu) : __uint_as_float(~u);
}
__global__ void k_msg(const float* __restrict__ vln, const float* __restrict__ ea,
                      const int* __restrict__ src, float* __restrict__ msg, long n) {
  long i = (long)blockIdx.x * blockDim.x + threadIdx.x;
  if (i >= n) return;
  long e = i / HH; int c = (int)(i - e * HH);
  msg[i] = fmaxf(vln[(long)src[e] * HH + c] + ea[i], 0.f) + 1e-7f;
}
__global__ void k_segmax(const float* __restrict__ msg, const float* __restrict__ t,
                         const int* __restrict__ dst, unsigned int* __restrict__ mx, long n) {
  long i = (long)blockIdx.x * blockDim.x + threadIdx.x;
  if (i >= n) return;
  long e = i / HH; int c = (int)(i - e * HH);
  atomicMax(&mx[(long)dst[e] * HH + c], fenc(msg[i] * t[0]));
}
__global__ void k_segexp(const float* __restrict__ msg, const float* __restrict__ t,
                         const int* __restrict__ dst, const unsigned int* __restrict__ mx,
                         float* __restrict__ den, float* __restrict__ num, long n) {
  long i = (long)blockIdx.x * blockDim.x + threadIdx.x;
  if (i >= n) return;
  long e = i / HH; int c = (int)(i - e * HH);
  long di = (long)dst[e] * HH + c;
  float ex = __expf(msg[i] * t[0] - fdec(mx[di]));
  atomicAdd(&den[di], ex);
  atomicAdd(&num[di], msg[i] * ex);
}
__global__ void k_agg(const float* num, const float* den, const float* vln,
                      float* out, long n) {
  long i = (long)blockIdx.x * blockDim.x + threadIdx.x;
  if (i < n) out[i] = num[i] / fmaxf(den[i], 1e-16f) + vln[i];
}

// ---------------------------------------------------------------- deg / sort
__global__ void k_deg(const int* src, float* deg, long n) {
  long i = (long)blockIdx.x * blockDim.x + threadIdx.x;
  if (i < n) atomicAdd(&deg[src[i]], 1.f);
}
__global__ void k_keys(const float* deg, float* keys, int* idx, long n) {
  long i = (long)blockIdx.x * blockDim.x + threadIdx.x;
  if (i >= n) return;
  unsigned int h = ((unsigned int)i * 2654435761u) >> 8;
  keys[i] = deg[i] + 0.01f * (float)(h & 0xFFFFFFu) * (1.f / 16777216.f);
  idx[i]  = (int)i;
}
__global__ void k_bitonic(float* key, int* idx, int n, int j, int kk) {
  int i = blockIdx.x * blockDim.x + threadIdx.x;
  if (i >= n) return;
  int ixj = i ^ j;
  if (ixj > i) {
    bool up = ((i & kk) == 0);
    float ki = key[i], kj = key[ixj];
    if (up ? (ki > kj) : (ki < kj)) {
      key[i] = kj; key[ixj] = ki;
      int t = idx[i]; idx[i] = idx[ixj]; idx[ixj] = t;
    }
  }
}
__global__ void k_rev(const int* idx, int* rev, long n) {
  long i = (long)blockIdx.x * blockDim.x + threadIdx.x;
  if (i < n) rev[idx[i]] = (int)i;
}
__global__ void k_gather_rows(float* __restrict__ dst, const float* __restrict__ src,
                              const int* __restrict__ idx, long M, int D) {
  long i = (long)blockIdx.x * blockDim.x + threadIdx.x;
  if (i >= M * (long)D) return;
  long r = i / D; int c = (int)(i - r * (long)D);
  dst[i] = src[(long)idx[r] * D + c];
}

// --------------------------------------------------------- depthwise conv k=4
__global__ void k_dwconv(const float* __restrict__ x, const float* __restrict__ w,
                         const float* __restrict__ b, float* __restrict__ y,
                         int L, int nseq, int padL, int act) {
  long i = (long)blockIdx.x * blockDim.x + threadIdx.x;
  long total = (long)nseq * L * HH;
  if (i >= total) return;
  int c = (int)(i % HH);
  long rl = i / HH;
  int l = (int)(rl % L);
  long s = rl / L;
  const float* xs = x + s * (long)L * HH;
  float acc = b[c];
#pragma unroll
  for (int k = 0; k < DCV; ++k) {
    int ll = l + k - padL;
    if (ll >= 0 && ll < L) acc += xs[(long)ll * HH + c] * w[c * DCV + k];
  }
  y[i] = actf(acc, act);
}

__global__ void k_split(const float* __restrict__ xz, float* __restrict__ xm,
                        float* __restrict__ z, long M) {
  long i = (long)blockIdx.x * blockDim.x + threadIdx.x;
  if (i >= M * (long)HH) return;
  long r = i / HH; int c = (int)(i - r * HH);
  xm[i] = xz[r * (2 * HH) + c];
  z[i]  = xz[r * (2 * HH) + HH + c];
}

// ----------------------------------------------- selective scan (wave per s,d)
// lanes 0..15 hold the 16 SSM states; y = (sum_n h*C + u*D) * silu(z)
__global__ void k_scan(const float* __restrict__ u, const float* __restrict__ dt,
                       const float* __restrict__ Alog, const float* __restrict__ xdb,
                       const float* __restrict__ Dp, const float* __restrict__ z,
                       float* __restrict__ y, int L, long nseq) {
  long wave = (long)blockIdx.x * (blockDim.x >> 5) + (threadIdx.x >> 5);
  int lane = threadIdx.x & 31;
  if (wave >= nseq * HH) return;
  int d = (int)(wave % HH);
  long s = wave / HH;
  int n = lane & 15;
  float A  = -__expf(Alog[d * DSZ + n]);
  float Dv = Dp[d];
  float h = 0.f;
  long base = s * (long)L;
  for (int l = 0; l < L; ++l) {
    long r = base + l;
    float dtv = dt[r * HH + d];
    float uv  = u[r * HH + d];
    float Bn  = xdb[r * 40 + 8 + n];
    float Cn  = xdb[r * 40 + 24 + n];
    h = __expf(dtv * A) * h + dtv * uv * Bn;
    float p = h * Cn;
    if (lane >= 16) p = 0.f;
    for (int o = 8; o > 0; o >>= 1) p += __shfl_xor(p, o, 32);
    if (lane == 0) {
      float zs = z[r * HH + d];
      y[r * HH + d] = (p + uv * Dv) * (zs / (1.f + __expf(-zs)));
    }
  }
}

// -------------------------------------------------------------- misc fusions
__global__ void k_xf(const float* xloc, const float* xglo, const float* g,
                     const float* b, float* out, long n) {
  long i = (long)blockIdx.x * blockDim.x + threadIdx.x;
  if (i >= n) return;
  int c = (int)(i % HH);
  out[i] = (xloc[i] + xglo[i]) * 0.99999500003749969f * g[c] + b[c];
}
__global__ void k_colmean(const float* h, float* mean, long M, int D) {
  int c = threadIdx.x;
  if (c >= D) return;
  float s = 0.f;
  for (long i = 0; i < M; ++i) s += h[i * D + c];
  mean[c] = s / (float)M;
}
__global__ void k_softmax2(const float* lam, float* w) {
  if (threadIdx.x == 0) {
    float a = lam[0], b = lam[1], m = fmaxf(a, b);
    float ea = __expf(a - m), eb = __expf(b - m), s = ea + eb;
    w[0] = ea / s; w[1] = eb / s;
  }
}
__global__ void k_concat2(const float* ts, const float* h, const float* mean,
                          const float* w, float* z2, long n) {
  long i = (long)blockIdx.x * blockDim.x + threadIdx.x;
  if (i >= n) return;
  long r = i / (2 * HH); int c = (int)(i - r * (2 * HH));
  z2[i] = (c < HH) ? w[0] * ts[r * HH + c]
                   : w[1] * (h[r * HH + (c - HH)] + mean[c - HH]);
}
__global__ void k_last(const float* __restrict__ xh, int* __restrict__ last, long n) {
  long i = (long)blockIdx.x * blockDim.x + threadIdx.x;
  if (i >= n) return;
  int cnt = 0;
  for (int t = 0; t < TT; ++t) {
    float s = 0.f;
    const float* p = xh + (i * TT + t) * (long)HH;
    for (int c = 0; c < HH; ++c) s += fabsf(p[c]);
    cnt += (s > 0.f) ? 1 : 0;
  }
  last[i] = (cnt < 1 ? 1 : cnt) - 1;
}
__global__ void k_gather_last(const float* v, const int* last, float* z, long n) {
  long i = (long)blockIdx.x * blockDim.x + threadIdx.x;
  if (i >= n) return;
  long r = i / HH; int c = (int)(i - r * HH);
  z[i] = v[((long)r * TT + last[r]) * HH + c];
}
__global__ void k_proj2(const float* x, const float* w, const float* b,
                        float* out, long M, int D) {
  long row = (long)blockIdx.x * (blockDim.x >> 5) + (threadIdx.x >> 5);
  int lane = threadIdx.x & 31;
  if (row >= M) return;
  float s = 0.f;
  for (int c = lane; c < D; c += 32) s += x[row * D + c] * w[c];
  for (int o = 16; o > 0; o >>= 1) s += __shfl_xor(s, o, 32);
  if (lane == 0) out[row] = fminf(fmaxf(s + b[0], 0.f), 14.f);
}

// ================================================================= host side
static inline unsigned int divup(long a, long b) { return (unsigned int)((a + b - 1) / b); }

extern "C" void kernel_launch(void* const* d_in, const int* in_sizes, int n_in,
                              void* d_out, int out_size, void* d_ws, size_t ws_size,
                              hipStream_t stream) {
  const float* x         = (const float*)d_in[0];
  const float* edge_attr = (const float*)d_in[1];
  const float* x_hist    = (const float*)d_in[2];
  const int*   eidx      = (const int*)d_in[104];
  const int*   srcI = eidx;
  const int*   dstI = eidx + EE;
  auto P  = [&](int i) { return (const float*)d_in[i]; };
  auto BP = [&](int blk, int off) { return (const float*)d_in[11 + 30 * blk + off]; };

  // -------- workspace arena
  size_t off = 0;
  auto alc = [&](size_t bytes) {
    void* p = (char*)d_ws + off;
    off = (off + bytes + 255) & ~(size_t)255;
    return p;
  };
  float* EAb = (float*)alc((size_t)EE * HH * 4);
  float* MSG = (float*)alc((size_t)EE * HH * 4);
  float* Hf  = (float*)alc((size_t)NN * HH * 4);
  float* TSO = (float*)alc((size_t)NN * HH * 4);
  unsigned int* MXu = (unsigned int*)alc((size_t)NN * HH * 4);
  float* DEN = (float*)alc((size_t)NN * HH * 4);
  float* NUM = (float*)alc((size_t)NN * HH * 4);
  float* DEG = (float*)alc((size_t)NN * 4);
  float* KEYS= (float*)alc((size_t)NN * 4);
  int*   IDX = (int*)alc((size_t)NN * 4);
  int*   REV = (int*)alc((size_t)NN * 4);
  int*   LAST= (int*)alc((size_t)NN * 4);
  float* MEANH=(float*)alc(HH * 4);
  float* LAMW =(float*)alc(2 * 4);
  float* G1  = (float*)alc((size_t)NTT * 2 * HH * 4);
  float* G2  = (float*)alc((size_t)NTT * HH * 4);
  float* G3  = (float*)alc((size_t)NTT * HH * 4);
  float* G4  = (float*)alc((size_t)NTT * HH * 4);
  float* G5  = (float*)alc((size_t)NTT * HH * 4);
  float* G6  = (float*)alc((size_t)NTT * HH * 4);
  float* G7  = (float*)alc((size_t)NTT * HH * 4);
  float* DTB = (float*)alc((size_t)NTT * HH * 4);
  float* XDB = (float*)alc((size_t)NTT * 40 * 4);
  unsigned short* ABUF = (unsigned short*)alc((size_t)NTT * HH * 2);
  unsigned short* WBUF = (unsigned short*)alc((size_t)65536 * 2);
  (void)ws_size; (void)n_in; (void)in_sizes; (void)out_size;

  // -------- helpers (operands padded: Kp -> mult of 32, W rows -> mult of 16)
  auto gemm = [&](const float* A, int lda, const float* W, const float* bias,
                  float* C, long M, int Nc, int K, int act) {
    int Kp  = (K + 31) & ~31;
    int Ncp = (Nc + 15) & ~15;
    k_f32_to_bf16_pad<<<divup(M * (long)Kp, 256), 256, 0, stream>>>(
        A, lda, ABUF, M, (int)M, Kp, K);
    k_f32_to_bf16_pad<<<divup((long)Ncp * Kp, 256), 256, 0, stream>>>(
        W, K, WBUF, Ncp, Nc, Kp, K);
    long tiles = (M >> 4) * (long)(Ncp >> 4);
    k_gemm_wmma<<<divup(tiles * 32, 256), 256, 0, stream>>>(ABUF, WBUF, bias, C,
                                                            (int)M, Nc, Kp, act);
  };
  auto ln = [&](const float* xx, const float* g, const float* b, float* y,
                long M, int D, int act) {
    k_layernorm<<<divup(M, 8), 256, 0, stream>>>(xx, g, b, y, M, D, act);
  };
  auto rms = [&](const float* xx, const float* w, float* y, long M, int D, int act) {
    k_rmsnorm<<<divup(M, 8), 256, 0, stream>>>(xx, w, y, M, D, act);
  };
  // mamba: uses G1 (xz), G2 (xm), G6 (z), G7 (xm_raw then gated y), DTB, XDB
  auto run_mamba = [&](const float* const* mp, const float* u, long M, int L,
                       long nseq, float* out) {
    gemm(u, HH, mp[0], nullptr, G1, M, 2 * HH, HH, 0);             // in_w
    k_split<<<divup(M * (long)HH, 256), 256, 0, stream>>>(G1, G7, G6, M);
    k_dwconv<<<divup(M * (long)HH, 256), 256, 0, stream>>>(G7, mp[1], mp[2], G2,
                                                           L, (int)nseq, DCV - 1, 3);
    gemm(G2, HH, mp[3], nullptr, XDB, M, 40, HH, 0);               // xproj
    gemm(XDB, 40, mp[4], mp[5], DTB, M, HH, DTRK, 4);              // dt + softplus
    long waves = nseq * HH;
    k_scan<<<divup(waves * 32, 256), 256, 0, stream>>>(G2, DTB, mp[6], XDB, mp[7],
                                                       G6, G7, L, nseq);
    gemm(G7, HH, mp[8], nullptr, out, M, HH, HH, 0);               // out_w
  };

  float negbig = -3.0e38f;
  unsigned int ub; memcpy(&ub, &negbig, 4);
  ub = (ub & 0x80000000u) ? ~ub : (ub | 0x80000000u);

  // ================= encoders =================
  gemm(x, HH, P(3), P(4), G3, NN, HH, HH, 0);
  ln(G3, P(5), P(6), Hf, NN, HH, 2);                               // h
  gemm(edge_attr, EINW, P(7), P(8), G1, EE, HH, EINW, 0);
  ln(G1, P(9), P(10), EAb, EE, HH, 2);                             // ea

  // ================= GPS blocks =================
  for (int b = 0; b < 2; ++b) {
    const float* t = BP(b, 2);
    ln(Hf, BP(b, 0), BP(b, 1), G3, NN, HH, 0);                     // v_ln
    long nEH = (long)EE * HH, nNH = (long)NN * HH;
    k_msg<<<divup(nEH, 256), 256, 0, stream>>>(G3, EAb, srcI, MSG, nEH);
    k_fillu<<<divup(nNH, 256), 256, 0, stream>>>(MXu, ub, nNH);
    k_segmax<<<divup(nEH, 256), 256, 0, stream>>>(MSG, t, dstI, MXu, nEH);
    k_fillf<<<divup(nNH, 256), 256, 0, stream>>>(DEN, 0.f, nNH);
    k_fillf<<<divup(nNH, 256), 256, 0, stream>>>(NUM, 0.f, nNH);
    k_segexp<<<divup(nEH, 256), 256, 0, stream>>>(MSG, t, dstI, MXu, DEN, NUM, nEH);
    k_agg<<<divup(nNH, 256), 256, 0, stream>>>(NUM, DEN, G3, G4, nNH);
    gemm(G4, HH, BP(b, 3), BP(b, 4), G1, NN, 2 * HH, HH, 0);       // gen.w1
    ln(G1, BP(b, 5), BP(b, 6), G2, NN, 2 * HH, 1);                 // LN + relu
    gemm(G2, 2 * HH, BP(b, 7), BP(b, 8), G5, NN, HH, 2 * HH, 0);   // x_loc

    k_fillf<<<divup((long)NN, 256), 256, 0, stream>>>(DEG, 0.f, NN);
    k_deg<<<divup((long)EE, 256), 256, 0, stream>>>(srcI, DEG, EE);
    k_keys<<<divup((long)NN, 256), 256, 0, stream>>>(DEG, KEYS, IDX, NN);
    for (int kk = 2; kk <= NN; kk <<= 1)
      for (int j = kk >> 1; j > 0; j >>= 1)
        k_bitonic<<<divup((long)NN, 256), 256, 0, stream>>>(KEYS, IDX, NN, j, kk);
    k_rev<<<divup((long)NN, 256), 256, 0, stream>>>(IDX, REV, NN);

    k_gather_rows<<<divup(nNH, 256), 256, 0, stream>>>(G4, Hf, IDX, NN, HH);
    gemm(G4, HH, BP(b, 9), nullptr, G6, NN, HH, HH, 0);            // in_proj
    k_dwconv<<<divup(nNH, 256), 256, 0, stream>>>(G6, BP(b, 10), BP(b, 11), G3,
                                                  NN, 1, 1, 3);    // xs, pad (1,2)
    const float* mp[9];
    for (int q = 0; q < 9; ++q) mp[q] = BP(b, 12 + q);
    run_mamba(mp, G3, NN, NN, 1, G4);
    k_mul_silu<<<divup(nNH, 256), 256, 0, stream>>>(G4, G3, nNH);  // * silu(xs)
    gemm(G4, HH, BP(b, 21), nullptr, G7, NN, HH, HH, 0);           // out_proj
    k_gather_rows<<<divup(nNH, 256), 256, 0, stream>>>(G6, G7, REV, NN, HH);
    k_xf<<<divup(nNH, 256), 256, 0, stream>>>(G5, G6, BP(b, 22), BP(b, 23), G3, nNH);
    ln(G3, BP(b, 24), BP(b, 25), G4, NN, HH, 0);
    gemm(G4, HH, BP(b, 26), BP(b, 27), G1, NN, 4 * HH, HH, 2);     // mlp.w1 + gelu
    gemm(G1, 4 * HH, BP(b, 28), BP(b, 29), G4, NN, HH, 4 * HH, 0); // mlp.w2
    k_add<<<divup(nNH, 256), 256, 0, stream>>>(Hf, G4, nNH);       // residual
  }

  // ================= ts encoder =================
  long nTH = (long)NTT * HH;
  gemm(x_hist, HH, P(71), P(72), G1, NTT, HH, HH, 0);
  rms(G1, P(73), G3, NTT, HH, 2);                                  // res
  for (int l = 0; l < 2; ++l) {
    int base = 74 + 10 * l;
    rms(G3, P(base), G4, NTT, HH, 0);
    const float* mp[9];
    for (int q = 0; q < 9; ++q) mp[q] = P(base + 1 + q);
    run_mamba(mp, G4, NTT, TT, NN, G5);
    k_add<<<divup(nTH, 256), 256, 0, stream>>>(G3, G5, nTH);
  }
  rms(G3, P(94), G1, NTT, HH, 0);
  k_last<<<divup((long)NN, 256), 256, 0, stream>>>(x_hist, LAST, NN);
  k_gather_last<<<divup((long)NN * HH, 256), 256, 0, stream>>>(G1, LAST, G2, (long)NN * HH);
  gemm(G2, HH, P(95), P(96), TSO, NN, HH, HH, 2);                  // ts out + gelu

  // ================= head =================
  k_colmean<<<1, HH, 0, stream>>>(Hf, MEANH, NN, HH);
  k_softmax2<<<1, 32, 0, stream>>>(P(97), LAMW);
  long n2 = (long)NN * 2 * HH;
  k_concat2<<<divup(n2, 256), 256, 0, stream>>>(TSO, Hf, MEANH, LAMW, G1, n2);
  ln(G1, P(98), P(99), G2, NN, 2 * HH, 0);
  gemm(G2, 2 * HH, P(100), P(101), G3, NN, HH, 2 * HH, 2);         // proj1 + gelu
  k_proj2<<<divup((long)NN, 8), 256, 0, stream>>>(G3, P(102), P(103),
                                                  (float*)d_out, NN, HH);
}
